// StandardAttention_24713241822284
// MI455X (gfx1250) — matrix operand order repaired and checked by reading, not compiled, not run on hardware
//
#include <hip/hip_runtime.h>

// MI455X / gfx1250, wave32. GEMMs via v_wmma_f32_16x16x32_bf16; depthwise conv
// staged through LDS with gfx1250 async global->LDS loads when available.
//
// Shapes: b=4, c=96, head=3, hc=32, h=w=256, n=65536.
// Workspace layout (bytes):
//   qkv1  bf16 [4][288][65536]  150,994,944   (after 1x1 conv)
//   qkv2  bf16 [4][288][65536]  150,994,944   (after dw conv; q|k|v = ch 0-95|96-191|192-287)
//   sumsq f32  [4][192]               3,072
//   S     f32  [4][3][32][32]        49,152   (raw Gram, split-K atomic accum)
//   M     bf16 [4][96][96]           73,728   (proj_w @ blockdiag(attn))
//   wbf   bf16 [288][96]             55,296   (qkv_w in bf16)

typedef __attribute__((ext_vector_type(16))) __bf16 v16bf;
typedef __attribute__((ext_vector_type(8)))  float  v8f;

#define NPIX  65536
#define CIN   96
#define C3    288

#if __has_builtin(__builtin_amdgcn_global_load_async_to_lds_b32) && \
    __has_builtin(__builtin_amdgcn_s_wait_asynccnt)
#define HAVE_ASYNC_LDS 1
#else
#define HAVE_ASYNC_LDS 0
#endif

// ---- WMMA lane-layout helpers (ISA 7.12.2, wave32) -------------------------
// 16-bit A 16x32: lane L: row M = L%16; half = L/16.
//   element j (0..15): K = (j<8 ? half*8 + j : 16 + half*8 + (j-8))
__device__ __forceinline__ int kA(int j, int half) {
  return (j < 8) ? (half * 8 + j) : (16 + half * 8 + (j - 8));
}

// A tile from bf16 row-major [M x ld]; per-lane elements are one contiguous
// 16x-bf16 run (K = half*16 + j) -> global_load_b128 pairs.
__device__ __forceinline__ v16bf loadA_bf16(const __bf16* __restrict__ src, size_t ld,
                                            int m0, int k0) {
  int lane = threadIdx.x & 31, half = lane >> 4, row = lane & 15;
  const __bf16* p = src + (size_t)(m0 + row) * ld + k0;
  v16bf a;
#pragma unroll
  for (int j = 0; j < 16; ++j) a[j] = p[half * 16 + j];
  return a;
}

// B 32x16 as loaded here: lane L: col N = L%16; half = L/16; element j: K = half*16 + j (same order as loadA_bf16).
// B from K-major memory (element (k,n) at k*ld + n), fp32 source.
__device__ __forceinline__ v16bf loadB_km_f32(const float* __restrict__ src, size_t ld,
                                              int k0, int n0) {
  int lane = threadIdx.x & 31, half = lane >> 4, col = lane & 15;
  const float* p = src + (size_t)(k0 + half * 16) * ld + n0 + col;
  v16bf r;
#pragma unroll
  for (int j = 0; j < 16; ++j) r[j] = (__bf16)p[(size_t)j * ld];
  return r;
}

// B from K-major memory, bf16 source.
__device__ __forceinline__ v16bf loadB_km_bf16(const __bf16* __restrict__ src, size_t ld,
                                               int k0, int n0) {
  int lane = threadIdx.x & 31, half = lane >> 4, col = lane & 15;
  const __bf16* p = src + (size_t)(k0 + half * 16) * ld + n0 + col;
  v16bf r;
#pragma unroll
  for (int j = 0; j < 16; ++j) r[j] = p[(size_t)j * ld];
  return r;
}

// B from N-major memory (element (k,n) at n*ld + k): K contiguous per lane.
__device__ __forceinline__ v16bf loadB_nm_bf16(const __bf16* __restrict__ src, size_t ld,
                                               int k0, int n0) {
  int lane = threadIdx.x & 31, half = lane >> 4, col = lane & 15;
  const __bf16* p = src + (size_t)(n0 + col) * ld + k0 + half * 16;
  v16bf r;
#pragma unroll
  for (int j = 0; j < 16; ++j) r[j] = p[j];
  return r;
}

__device__ __forceinline__ v8f wmma_bf16(v16bf a, v16bf b, v8f c) {
  return __builtin_amdgcn_wmma_f32_16x16x32_bf16(false, a, false, b, (short)0, c,
                                                 false, false);
}

// ---- K0: zero split-K accumulators ----------------------------------------
__global__ void k_zero(float* __restrict__ p, int n) {
  int i = blockIdx.x * blockDim.x + threadIdx.x;
  if (i < n) p[i] = 0.0f;
}

// ---- K0b: qkv_w fp32 -> bf16 ----------------------------------------------
__global__ void k_cvt_w(const float* __restrict__ w, __bf16* __restrict__ wbf, int n) {
  int i = blockIdx.x * blockDim.x + threadIdx.x;
  if (i < n) wbf[i] = (__bf16)w[i];
}

// ---- K1: qkv 1x1 conv as GEMM [288 x 96] x [96 x NPIX] per batch ----------
// grid (512 pixtiles, 4 b), block 256 = 8 waves; each wave owns 16 pixels,
// caches its 3 B (x) k-tiles in registers and sweeps all 18 M-tiles.
__global__ void k_qkv_pointwise(const float* __restrict__ x,
                                const __bf16* __restrict__ wbf,
                                const float* __restrict__ qkv_b,
                                __bf16* __restrict__ qkv1) {
  int wave = threadIdx.x >> 5, lane = threadIdx.x & 31;
  int pix0 = blockIdx.x * 128 + wave * 16;
  int b = blockIdx.y;
  const float* xb = x + (size_t)b * CIN * NPIX;
  v16bf bt0 = loadB_km_f32(xb, NPIX, 0,  pix0);
  v16bf bt1 = loadB_km_f32(xb, NPIX, 32, pix0);
  v16bf bt2 = loadB_km_f32(xb, NPIX, 64, pix0);
  int half = lane >> 4, col = lane & 15;
  size_t base = (size_t)b * C3 * NPIX;
  for (int mt = 0; mt < 18; ++mt) {
    int m0 = mt * 16;
    v8f acc = {};
    acc = wmma_bf16(loadA_bf16(wbf, CIN, m0, 0),  bt0, acc);
    acc = wmma_bf16(loadA_bf16(wbf, CIN, m0, 32), bt1, acc);
    acc = wmma_bf16(loadA_bf16(wbf, CIN, m0, 64), bt2, acc);
#pragma unroll
    for (int r = 0; r < 8; ++r) {
      int m = m0 + r + half * 8;              // C/D: VGPR r -> M=r / r+8
      qkv1[base + (size_t)m * NPIX + pix0 + col] = (__bf16)(acc[r] + qkv_b[m]);
    }
  }
}

// ---- K2: 3x3 depthwise conv + bias via LDS-staged rows; sumsq for q,k ------
// grid (256 rows, 288 ch, 4 b), block 256 threads = one output row
__global__ void k_dwconv(const __bf16* __restrict__ qkv1,
                         const float* __restrict__ dw_w,
                         const float* __restrict__ dw_b,
                         __bf16* __restrict__ qkv2,
                         float* __restrict__ sumsq) {
  int tid = threadIdx.x;
  int h = blockIdx.x, ch = blockIdx.y, b = blockIdx.z;
  const __bf16* src = qkv1 + ((size_t)(b * C3 + ch)) * NPIX;

  __shared__ unsigned int tilew[3 * 128];            // 3 rows x 256 bf16
  const __bf16* tile = (const __bf16*)tilew;

  // stage rows h-1..h+1 (384 dwords); out-of-image rows zero-filled
  for (int w = tid; w < 384; w += 256) {
    int r = w >> 7, cw = w & 127;
    int hh = h - 1 + r;
    if (hh >= 0 && hh < 256) {
#if HAVE_ASYNC_LDS
      __builtin_amdgcn_global_load_async_to_lds_b32(
          (__attribute__((address_space(1))) int*)(src + hh * 256 + cw * 2),
          (__attribute__((address_space(3))) int*)&tilew[w], 0, 0);
#else
      tilew[w] = ((const unsigned int*)(src + (size_t)hh * 256))[cw];
#endif
    } else {
      tilew[w] = 0u;
    }
  }
#if HAVE_ASYNC_LDS
  __builtin_amdgcn_s_wait_asynccnt(0);
#endif
  __syncthreads();

  float wgt[9];
#pragma unroll
  for (int i = 0; i < 9; ++i) wgt[i] = dw_w[ch * 9 + i];
  float acc = dw_b[ch];
#pragma unroll
  for (int r = 0; r < 3; ++r) {
#pragma unroll
    for (int dx = -1; dx <= 1; ++dx) {
      int ww = tid + dx;
      if (ww >= 0 && ww < 256)
        acc += wgt[r * 3 + (dx + 1)] * (float)tile[r * 256 + ww];
    }
  }
  __bf16 stored = (__bf16)acc;
  qkv2[((size_t)(b * C3 + ch)) * NPIX + h * 256 + tid] = stored;

  if (ch < 192) {  // q or k channel: accumulate ||.||^2 on bf16-rounded values
    float sv = (float)stored;
    __shared__ float red[256];
    red[tid] = sv * sv;
    __syncthreads();
    for (int s = 128; s > 0; s >>= 1) {
      if (tid < s) red[tid] += red[tid + s];
      __syncthreads();
    }
    if (tid == 0) atomicAdd(&sumsq[b * 192 + ch], red[0]);
  }
}

// ---- K3: raw Gram S = q . k^T over n=65536, split-K WMMA -------------------
// grid (32 ksplit, 12 b*head), block 256 = 8 waves: tile t = wave%4 (2x2 of
// 16x16), ksub = wave/4 halves the 2048-pixel K slab.
__global__ void k_gram(const __bf16* __restrict__ qkv2, float* __restrict__ S) {
  int wave = threadIdx.x >> 5, lane = threadIdx.x & 31;
  int t = wave & 3, ksub = wave >> 2;
  int tm = t >> 1, tn = t & 1;
  int bh = blockIdx.y, b = bh / 3, head = bh % 3;
  const __bf16* qrow = qkv2 + ((size_t)(b * C3) + head * 32 + tm * 16) * NPIX;
  const __bf16* krow = qkv2 + ((size_t)(b * C3) + CIN + head * 32 + tn * 16) * NPIX;
  int p0 = blockIdx.x * 2048 + ksub * 1024;
  v8f acc = {};
#pragma unroll 4
  for (int kk = 0; kk < 1024; kk += 32) {
    __builtin_prefetch(qrow + p0 + kk + 2048, 0, 3);   // global_prefetch_b8
    __builtin_prefetch(krow + p0 + kk + 2048, 0, 3);
    v16bf a  = loadA_bf16(qrow, NPIX, 0, p0 + kk);     // A[c][p] = q
    v16bf bm = loadB_nm_bf16(krow, NPIX, p0 + kk, 0);  // B[p][d] = k^T
    acc = wmma_bf16(a, bm, acc);
  }
  int half = lane >> 4, col = lane & 15;
  float* Sbh = S + (size_t)bh * 1024;
#pragma unroll
  for (int r = 0; r < 8; ++r)
    atomicAdd(&Sbh[(tm * 16 + r + half * 8) * 32 + tn * 16 + col], acc[r]);
}

// ---- K4: scale by inv-norms, softmax -> attn output; build M ---------------
// grid 4 (b), block 256
__global__ void k_softmax_M(const float* __restrict__ S,
                            const float* __restrict__ sumsq,
                            const float* __restrict__ proj_w,
                            float* __restrict__ attn_out,
                            __bf16* __restrict__ M) {
  int b = blockIdx.x, tid = threadIdx.x;
  __shared__ float attn[3 * 32 * 32];
  if (tid < 96) {
    int head = tid / 32, c = tid % 32;
    float iq = 1.0f / fmaxf(sqrtf(sumsq[b * 192 + head * 32 + c]), 1e-12f);
    const float* Srow = S + ((size_t)(b * 3 + head)) * 1024 + c * 32;
    float vals[32], mx = -3.4e38f;
#pragma unroll
    for (int d = 0; d < 32; ++d) {
      float ik = 1.0f / fmaxf(sqrtf(sumsq[b * 192 + 96 + head * 32 + d]), 1e-12f);
      vals[d] = Srow[d] * iq * ik;   // TEMP = 1
      mx = fmaxf(mx, vals[d]);
    }
    float sum = 0.0f;
#pragma unroll
    for (int d = 0; d < 32; ++d) { vals[d] = __expf(vals[d] - mx); sum += vals[d]; }
    float inv = 1.0f / sum;
#pragma unroll
    for (int d = 0; d < 32; ++d) {
      float av = vals[d] * inv;
      attn[tid * 32 + d] = av;
      attn_out[(size_t)b * 3072 + tid * 32 + d] = av;
    }
  }
  __syncthreads();
  // M[o][head*32+d] = sum_m proj_w[o][head*32+m] * attn[head][m][d]
  for (int idx = tid; idx < 96 * 96; idx += 256) {
    int o = idx / 96, cc = idx % 96;
    int head = cc / 32, d = cc % 32;
    float s = 0.0f;
#pragma unroll
    for (int m = 0; m < 32; ++m)
      s += proj_w[o * 96 + head * 32 + m] * attn[(head * 32 + m) * 32 + d];
    M[(size_t)b * 9216 + idx] = (__bf16)s;
  }
}

// ---- K5: out = M @ v + proj_b (fused attn.v + proj), WMMA ------------------
// grid (512 pixtiles, 4 b), block 256 = 8 waves; B tiles cached in registers,
// sweep the 6 M-tiles.
__global__ void k_proj(const __bf16* __restrict__ M,
                       const __bf16* __restrict__ qkv2,
                       const float* __restrict__ proj_b,
                       float* __restrict__ out) {
  int wave = threadIdx.x >> 5, lane = threadIdx.x & 31;
  int pix0 = blockIdx.x * 128 + wave * 16;
  int b = blockIdx.y;
  const __bf16* Mb = M + (size_t)b * 9216;
  const __bf16* vb = qkv2 + ((size_t)b * C3 + 192) * NPIX;  // v channels
  v16bf bt0 = loadB_km_bf16(vb, NPIX, 0,  pix0);
  v16bf bt1 = loadB_km_bf16(vb, NPIX, 32, pix0);
  v16bf bt2 = loadB_km_bf16(vb, NPIX, 64, pix0);
  int half = lane >> 4, col = lane & 15;
  for (int mt = 0; mt < 6; ++mt) {
    int m0 = mt * 16;
    v8f acc = {};
    acc = wmma_bf16(loadA_bf16(Mb, CIN, m0, 0),  bt0, acc);
    acc = wmma_bf16(loadA_bf16(Mb, CIN, m0, 32), bt1, acc);
    acc = wmma_bf16(loadA_bf16(Mb, CIN, m0, 64), bt2, acc);
#pragma unroll
    for (int r = 0; r < 8; ++r) {
      int m = m0 + r + half * 8;
      out[((size_t)b * CIN + m) * NPIX + pix0 + col] = acc[r] + proj_b[m];
    }
  }
}

extern "C" void kernel_launch(void* const* d_in, const int* in_sizes, int n_in,
                              void* d_out, int out_size, void* d_ws, size_t ws_size,
                              hipStream_t stream) {
  const float* x      = (const float*)d_in[0];
  const float* qkv_w  = (const float*)d_in[1];
  const float* qkv_b  = (const float*)d_in[2];
  const float* dw_w   = (const float*)d_in[3];
  const float* dw_b   = (const float*)d_in[4];
  const float* proj_w = (const float*)d_in[5];
  const float* proj_b = (const float*)d_in[6];

  float* out      = (float*)d_out;
  float* attn_out = out + (size_t)4 * CIN * NPIX;  // second tuple element

  char* ws = (char*)d_ws;
  const size_t SZ_QKV = (size_t)4 * C3 * NPIX * sizeof(__bf16);  // 150,994,944
  __bf16* qkv1  = (__bf16*)(ws);
  __bf16* qkv2  = (__bf16*)(ws + SZ_QKV);
  float*  sumsq = (float*)(ws + 2 * SZ_QKV);                       // 768 f
  float*  S     = (float*)(ws + 2 * SZ_QKV + 3072);                // 12288 f
  __bf16* Mmat  = (__bf16*)(ws + 2 * SZ_QKV + 3072 + 49152);       // 9216*4
  __bf16* wbf   = (__bf16*)(ws + 2 * SZ_QKV + 3072 + 49152 + 73728);

  // zero sumsq + S (contiguous: 768 + 12288 = 13056 floats)
  k_zero<<<dim3((13056 + 255) / 256), 256, 0, stream>>>(sumsq, 13056);
  // qkv_w -> bf16
  k_cvt_w<<<dim3((C3 * CIN + 255) / 256), 256, 0, stream>>>(qkv_w, wbf, C3 * CIN);
  // 1x1 qkv conv (WMMA bf16, B cached in registers across M-tiles)
  k_qkv_pointwise<<<dim3(512, 4), 256, 0, stream>>>(x, wbf, qkv_b, qkv1);
  // depthwise 3x3 (LDS-staged, async global->LDS) + per-channel sum-of-squares
  k_dwconv<<<dim3(256, C3, 4), 256, 0, stream>>>(qkv1, dw_w, dw_b, qkv2, sumsq);
  // raw Gram matrices (WMMA bf16, split-K atomics)
  k_gram<<<dim3(32, 12), 256, 0, stream>>>(qkv2, S);
  // normalize-scale + softmax + M = proj_w @ blockdiag(attn)
  k_softmax_M<<<dim3(4), 256, 0, stream>>>(S, sumsq, proj_w, attn_out, Mmat);
  // fused (attn @ v) + proj as one WMMA GEMM
  k_proj<<<dim3(512, 4), 256, 0, stream>>>(Mmat, qkv2, proj_b, out);
}
